// SparseAttention_16784732193087
// MI455X (gfx1250) — matrix operand order repaired and checked by reading, not compiled, not run on hardware
//
#include <hip/hip_runtime.h>

// ---------------------------------------------------------------------------
// Problem constants (match reference): B=2, S=2048, D=1024, H=16, K=16
// ---------------------------------------------------------------------------
#define B_ 2
#define S_ 2048
#define D_ 1024
#define H_ 16
#define DH_ 64          // head dim
#define HID_ 512        // D/2 indexer hidden
#define TOPK_ 16
#define N_ (B_ * S_)    // 4096 total rows

typedef unsigned short u16;
typedef __attribute__((ext_vector_type(2)))  unsigned int u32x2;
typedef __attribute__((ext_vector_type(4)))  unsigned int u32x4;
typedef __attribute__((ext_vector_type(4)))  float        f32x4;
typedef __attribute__((ext_vector_type(8)))  float        v8f;
typedef __attribute__((ext_vector_type(16))) __bf16       v16bf;

union FragAB { u32x4 q[2]; v16bf v; };
union H8     { u32x4 q;    u16   s[8]; };

__device__ __forceinline__ u16 f2bf(float f) {
  union { float f; unsigned u; } a; a.f = f;
  unsigned r = a.u + 0x7FFFu + ((a.u >> 16) & 1u);
  return (u16)(r >> 16);
}
__device__ __forceinline__ float bf2f(u16 h) {
  union { unsigned u; float f; } a; a.u = ((unsigned)h) << 16;
  return a.f;
}
__device__ __forceinline__ v8f zero8() {
  v8f z = {0.f, 0.f, 0.f, 0.f, 0.f, 0.f, 0.f, 0.f};
  return z;
}
// 16 bf16 halves from two (possibly non-adjacent) 16B-aligned groups of 8
__device__ __forceinline__ v16bf load_frag16(const u16* p0, const u16* p1) {
  FragAB f;
  f.q[0] = *(const u32x4*)p0;
  f.q[1] = *(const u32x4*)p1;
  return f.v;
}
__device__ __forceinline__ v8f wmma_bf16(v16bf a, v16bf b, v8f c) {
  // D(f32 16x16) = A(bf16 16x32) * B(bf16 32x16) + C
  return __builtin_amdgcn_wmma_f32_16x16x32_bf16(false, a, false, b,
                                                 (short)0, c, false, false);
}

// ---------------------------------------------------------------------------
// CDNA5 async memory->LDS copy (ASYNCcnt-tracked, ISA §10.7 / §15.18).
// The low 32 bits of a flat __shared__ pointer are the LDS byte address
// (ISA §10.2: LDS aperture maps addr[31:0] -> LDS_ADDR).
// ---------------------------------------------------------------------------
__device__ __forceinline__ void async_ld_b128(void* lds, const void* gptr) {
  unsigned loff = (unsigned)(uintptr_t)lds;
  asm volatile("global_load_async_to_lds_b128 %0, %1, off"
               :: "v"(loff), "v"(gptr) : "memory");
}
__device__ __forceinline__ void wait_asynccnt0() {
  asm volatile("s_wait_asynccnt 0x0" ::: "memory");
}

// ---------------------------------------------------------------------------
// Kernel 1: tiled WMMA GEMM  C[M,N] = A[M,K] @ Bw[K,N] + bias  (opt ReLU)
//   A: fp32 or bf16 (A_BF16), row-major.  Bw: fp32 row-major.
//   C: bf16 (OUT_BF16) or fp32.
//   Block: 128 threads (4 waves).  Tile: 128x64, BK=32; wave strip 32x64
//   (2 A-frags x 4 B-frags = 8 WMMA per staged tile, B-frag reused twice).
// ---------------------------------------------------------------------------
template <bool A_BF16, bool OUT_BF16, bool RELU>
__global__ __launch_bounds__(128) void gemm_wmma_kernel(
    const void* __restrict__ Av, const float* __restrict__ Bw,
    const float* __restrict__ bias, void* __restrict__ Cv,
    int M, int Nn, int Kk) {
  __shared__ u16 As[128][32];   // A tile, bf16 row-major [m][k]   (8 KB)
  __shared__ u16 Bts[64][32];   // B tile, bf16 TRANSPOSED [n][k]  (4 KB)

  const int t = threadIdx.x;
  const int lane = t & 31, w = t >> 5;
  const int ln = lane & 15, rsel = lane >> 4;
  const int rowBase = blockIdx.y * 128;
  const int colBase = blockIdx.x * 64;

  v8f acc[2][4] = {{zero8(), zero8(), zero8(), zero8()},
                   {zero8(), zero8(), zero8(), zero8()}};

  for (int kt = 0; kt < Kk; kt += 32) {
    __syncthreads();
    // ---- stage A tile (128x32) as bf16 ----
#pragma unroll
    for (int p = 0; p < 8; ++p) {
      const int e = (p * 128 + t) * 4;  // 0..4092 step 4
      const int r = e >> 5, c = e & 31;
      if (A_BF16) {
        const u16* Ag = (const u16*)Av;
        u32x2 v = *(const u32x2*)(Ag + (size_t)(rowBase + r) * Kk + kt + c);
        *(u32x2*)&As[r][c] = v;
      } else {
        const float* Ag = (const float*)Av;
        f32x4 v = *(const f32x4*)(Ag + (size_t)(rowBase + r) * Kk + kt + c);
        As[r][c + 0] = f2bf(v[0]); As[r][c + 1] = f2bf(v[1]);
        As[r][c + 2] = f2bf(v[2]); As[r][c + 3] = f2bf(v[3]);
      }
    }
    // ---- stage B tile (32x64) transposed to [n][k] as bf16 ----
#pragma unroll
    for (int p = 0; p < 4; ++p) {
      const int e = (p * 128 + t) * 4;  // over 32*64
      const int kk = e >> 6, nn = e & 63;
      f32x4 v = *(const f32x4*)(Bw + (size_t)(kt + kk) * Nn + colBase + nn);
      Bts[nn + 0][kk] = f2bf(v[0]); Bts[nn + 1][kk] = f2bf(v[1]);
      Bts[nn + 2][kk] = f2bf(v[2]); Bts[nn + 3][kk] = f2bf(v[3]);
    }
    if (kt + 32 < Kk) {  // hint next B tile into cache
      __builtin_prefetch(Bw + (size_t)(kt + 32 + (t >> 2)) * Nn + colBase, 0, 0);
    }
    __syncthreads();

    // ---- WMMA: each wave computes its 32x64 strip ----
    const u16* ap0 = &As[w * 32 + ln][0];
    const u16* ap1 = &As[w * 32 + 16 + ln][0];
    v16bf a0 = load_frag16(ap0 + rsel * 8, ap0 + 16 + rsel * 8);
    v16bf a1 = load_frag16(ap1 + rsel * 8, ap1 + 16 + rsel * 8);
#pragma unroll
    for (int n = 0; n < 4; ++n) {
      const u16* bp = &Bts[n * 16 + ln][rsel * 8];
      v16bf b = load_frag16(bp, bp + 16);
      acc[0][n] = wmma_bf16(a0, b, acc[0][n]);
      acc[1][n] = wmma_bf16(a1, b, acc[1][n]);
    }
  }

  // ---- epilogue: bias (+ReLU), store ----
#pragma unroll
  for (int s = 0; s < 2; ++s) {
    const int row0 = rowBase + w * 32 + s * 16 + rsel * 8;
#pragma unroll
    for (int n = 0; n < 4; ++n) {
      const int col = colBase + n * 16 + ln;
      const float bv = bias[col];
#pragma unroll
      for (int r = 0; r < 8; ++r) {
        float val = acc[s][n][r] + bv;
        if (RELU) val = val > 0.f ? val : 0.f;
        const size_t off = (size_t)(row0 + r) * Nn + col;
        if (OUT_BF16) ((u16*)Cv)[off] = f2bf(val);
        else          ((float*)Cv)[off] = val;
      }
    }
  }
}

// ---------------------------------------------------------------------------
// Kernel 2: importance GEMV  imp[i] = sigmoid(dot(Hb[i,:], Ws2) + bs2)
//   Hb: bf16 [N_, HID_].  One wave per row, 8 rows per 256-thread block.
// ---------------------------------------------------------------------------
__global__ __launch_bounds__(256) void imp_kernel(
    const u16* __restrict__ Hb, const float* __restrict__ Ws2,
    const float* __restrict__ bs2, float* __restrict__ imp) {
  const int lane = threadIdx.x & 31, w = threadIdx.x >> 5;
  const int row = blockIdx.x * 8 + w;
  float s = 0.f;
  for (int c = lane; c < HID_; c += 32)
    s += bf2f(Hb[(size_t)row * HID_ + c]) * Ws2[c];
#pragma unroll
  for (int off = 16; off; off >>= 1) s += __shfl_xor(s, off, 32);
  if (lane == 0) imp[row] = 1.f / (1.f + __expf(-(s + bs2[0])));
}

// ---------------------------------------------------------------------------
// Kernel 3: per-batch iterative top-K (K=16) -> sel mask (0/1 ints)
// ---------------------------------------------------------------------------
__global__ __launch_bounds__(256) void topk_kernel(
    const float* __restrict__ imp, int* __restrict__ sel) {
  __shared__ float vals[S_];
  __shared__ float rv[256];
  __shared__ int   ri[256];
  const int b = blockIdx.x, t = threadIdx.x;
  for (int i = t; i < S_; i += 256) {
    vals[i] = imp[b * S_ + i];
    sel[b * S_ + i] = 0;
  }
  __syncthreads();
  for (int it = 0; it < TOPK_; ++it) {
    float bv = -3.4e38f; int bi = 0;
    for (int i = t; i < S_; i += 256)
      if (vals[i] > bv) { bv = vals[i]; bi = i; }   // strict > : lowest index wins ties
    rv[t] = bv; ri[t] = bi;
    __syncthreads();
    for (int off = 128; off; off >>= 1) {
      if (t < off) {
        if (rv[t + off] > rv[t] ||
            (rv[t + off] == rv[t] && ri[t + off] < ri[t])) {
          rv[t] = rv[t + off]; ri[t] = ri[t + off];
        }
      }
      __syncthreads();
    }
    if (t == 0) { sel[b * S_ + ri[0]] = 1; vals[ri[0]] = -3.4e38f; }
    __syncthreads();
  }
}

// ---------------------------------------------------------------------------
// Kernel 4: flash-style masked attention for one (b,h), 16 queries per wave.
//   scores = (Q Kt)/8; mask kept if sel[q] || sel[k]; online softmax; O = P V.
//   Block: 128 threads (4 waves, 4 query tiles); key chunks of 32.
//   K chunk staged to LDS with ASYNC global->LDS copies (CDNA5 ASYNCcnt).
// ---------------------------------------------------------------------------
__global__ __launch_bounds__(128) void attn_kernel(
    const u16* __restrict__ Qb, const u16* __restrict__ Kb,
    const u16* __restrict__ Vb, const int* __restrict__ sel,
    u16* __restrict__ ATTb) {
  __shared__ u16 Ks[32][64];        // K chunk row-major [key][d]   (4 KB)
  __shared__ u16 Vt[64][32];        // V chunk transposed [d][key]  (4 KB)
  __shared__ u16 Pb[4][16][32];     // per-wave P scratch (C-layout -> A-layout)

  const int t = threadIdx.x, lane = t & 31, w = t >> 5;
  const int bh = blockIdx.y, b = bh >> 4, h = bh & (H_ - 1);
  const int qrow0 = blockIdx.x * 64 + w * 16;
  const int ln = lane & 15, rsel = lane >> 4, rbase = rsel * 8;

  // Q A-fragments for d 0..31 and 32..63 (held for the whole loop)
  const size_t rowQ = (size_t)(b * S_ + qrow0 + ln) * D_ + h * DH_;
  const v16bf qa0 = load_frag16(Qb + rowQ + rsel * 8,      Qb + rowQ + 16 + rsel * 8);
  const v16bf qa1 = load_frag16(Qb + rowQ + 32 + rsel * 8, Qb + rowQ + 48 + rsel * 8);

  // sel bits for this lane-half's 8 query rows
  unsigned qm = 0;
#pragma unroll
  for (int r = 0; r < 8; ++r)
    qm |= (sel[b * S_ + qrow0 + rbase + r] ? 1u : 0u) << r;

  v8f o0 = zero8(), o1 = zero8(), o2 = zero8(), o3 = zero8();
  float m[8], l[8];
#pragma unroll
  for (int r = 0; r < 8; ++r) { m[r] = -1e30f; l[r] = 0.f; }

  // per-thread async-copy assignments: 2 x 16B segments of the K chunk
  // segment si (0..255): key = si>>3, d = (si&7)*8
  for (int kb = 0; kb < S_; kb += 32) {
    __syncthreads();
    // ---- async-stage K chunk row-major into LDS (ASYNCcnt path) ----
#pragma unroll
    for (int i = 0; i < 2; ++i) {
      const int si = t + i * 128;
      const int key = si >> 3, dd = (si & 7) * 8;
      async_ld_b128(&Ks[key][dd],
                    Kb + (size_t)(b * S_ + kb + key) * D_ + h * DH_ + dd);
    }
    // ---- stage V chunk transposed into LDS: Vt[d][key] ----
    {
      const int j = t & 31, d0 = (t >> 5) * 16;
      const u16* vp = Vb + (size_t)(b * S_ + kb + j) * D_ + h * DH_ + d0;
      H8 v0; v0.q = *(const u32x4*)vp;
      H8 v1; v1.q = *(const u32x4*)(vp + 8);
#pragma unroll
      for (int i = 0; i < 8; ++i) {
        Vt[d0 + i][j]     = v0.s[i];
        Vt[d0 + 8 + i][j] = v1.s[i];
      }
    }
    wait_asynccnt0();     // this wave's async K copies have landed in LDS
    __syncthreads();      // => all waves' copies + Vt stores visible

    // ---- scores: two 16x16 tiles (keys kb..+15, kb+16..+31), Kdim=64 ----
    const u16* kp0 = &Ks[ln][rsel * 8];
    const u16* kp1 = &Ks[16 + ln][rsel * 8];
    v8f s0 = zero8(), s1 = zero8();
    s0 = wmma_bf16(qa0, load_frag16(kp0,      kp0 + 16), s0);
    s0 = wmma_bf16(qa1, load_frag16(kp0 + 32, kp0 + 48), s0);
    s1 = wmma_bf16(qa0, load_frag16(kp1,      kp1 + 16), s1);
    s1 = wmma_bf16(qa1, load_frag16(kp1 + 32, kp1 + 48), s1);

    const int sk0 = sel[b * S_ + kb + ln];
    const int sk1 = sel[b * S_ + kb + 16 + ln];

    // ---- mask + online softmax (row lives across the 16-lane half) ----
#pragma unroll
    for (int r = 0; r < 8; ++r) {
      float x0 = s0[r] * 0.125f, x1 = s1[r] * 0.125f;
      const bool qr = (qm >> r) & 1u;
      if (!(qr || sk0)) x0 = -3.0e38f;
      if (!(qr || sk1)) x1 = -3.0e38f;
      float rm = fmaxf(x0, x1);
      rm = fmaxf(rm, __shfl_xor(rm, 8, 16));
      rm = fmaxf(rm, __shfl_xor(rm, 4, 16));
      rm = fmaxf(rm, __shfl_xor(rm, 2, 16));
      rm = fmaxf(rm, __shfl_xor(rm, 1, 16));
      const float mn = fmaxf(m[r], rm);
      const float sc = __expf(m[r] - mn);
      const float p0 = __expf(x0 - mn);
      const float p1 = __expf(x1 - mn);
      float rs = p0 + p1;
      rs += __shfl_xor(rs, 8, 16);
      rs += __shfl_xor(rs, 4, 16);
      rs += __shfl_xor(rs, 2, 16);
      rs += __shfl_xor(rs, 1, 16);
      l[r] = l[r] * sc + rs;
      m[r] = mn;
      o0[r] *= sc; o1[r] *= sc; o2[r] *= sc; o3[r] *= sc;
      Pb[w][rbase + r][ln]      = f2bf(p0);
      Pb[w][rbase + r][16 + ln] = f2bf(p1);
    }

    // ---- PV: O(16x64) += P(16x32) @ Vchunk(32x64) ----
    const u16* pp = &Pb[w][ln][0];
    const v16bf pa = load_frag16(pp + rsel * 8, pp + 16 + rsel * 8);
    {
      const u16* v0 = &Vt[ 0 + ln][rsel * 8];
      const u16* v1 = &Vt[16 + ln][rsel * 8];
      const u16* v2 = &Vt[32 + ln][rsel * 8];
      const u16* v3 = &Vt[48 + ln][rsel * 8];
      o0 = wmma_bf16(pa, load_frag16(v0, v0 + 16), o0);
      o1 = wmma_bf16(pa, load_frag16(v1, v1 + 16), o1);
      o2 = wmma_bf16(pa, load_frag16(v2, v2 + 16), o2);
      o3 = wmma_bf16(pa, load_frag16(v3, v3 + 16), o3);
    }
  }

  // ---- finalize: divide by row sums, store bf16 [B,S,D] (head slice) ----
#pragma unroll
  for (int r = 0; r < 8; ++r) {
    const float li = 1.f / l[r];
    const size_t off = (size_t)(b * S_ + qrow0 + rbase + r) * D_ + h * DH_;
    ATTb[off +  0 + ln] = f2bf(o0[r] * li);
    ATTb[off + 16 + ln] = f2bf(o1[r] * li);
    ATTb[off + 32 + ln] = f2bf(o2[r] * li);
    ATTb[off + 48 + ln] = f2bf(o3[r] * li);
  }
}

// ---------------------------------------------------------------------------
// Host launcher
// ---------------------------------------------------------------------------
extern "C" void kernel_launch(void* const* d_in, const int* in_sizes, int n_in,
                              void* d_out, int out_size, void* d_ws, size_t ws_size,
                              hipStream_t stream) {
  (void)in_sizes; (void)n_in; (void)out_size; (void)ws_size;
  const float* x   = (const float*)d_in[0];
  const float* Wq  = (const float*)d_in[1];
  const float* bq  = (const float*)d_in[2];
  const float* Wk  = (const float*)d_in[3];
  const float* bk  = (const float*)d_in[4];
  const float* Wv  = (const float*)d_in[5];
  const float* bv  = (const float*)d_in[6];
  const float* Wo  = (const float*)d_in[7];
  const float* bo  = (const float*)d_in[8];
  const float* Ws1 = (const float*)d_in[9];
  const float* bs1 = (const float*)d_in[10];
  const float* Ws2 = (const float*)d_in[11];
  const float* bs2 = (const float*)d_in[12];

  // workspace carve-up (bf16 intermediates) — ~36 MB total
  u16* Qb   = (u16*)d_ws;
  u16* Kb   = Qb + (size_t)N_ * D_;
  u16* Vb   = Kb + (size_t)N_ * D_;
  u16* Hb   = Vb + (size_t)N_ * D_;           // [N_, HID_]
  u16* ATTb = Hb + (size_t)N_ * HID_;
  float* imp = (float*)(ATTb + (size_t)N_ * D_);
  int*   sel = (int*)(imp + N_);

  const dim3 blk(128);
  const dim3 gProj(D_ / 64, N_ / 128);    // (16, 32)
  const dim3 gHid(HID_ / 64, N_ / 128);   // (8, 32)

  // QKV projections (fp32 A -> bf16 out)
  gemm_wmma_kernel<false, true, false><<<gProj, blk, 0, stream>>>(x, Wq, bq, Qb, N_, D_, D_);
  gemm_wmma_kernel<false, true, false><<<gProj, blk, 0, stream>>>(x, Wk, bk, Kb, N_, D_, D_);
  gemm_wmma_kernel<false, true, false><<<gProj, blk, 0, stream>>>(x, Wv, bv, Vb, N_, D_, D_);

  // indexer hidden: relu(x @ Ws1 + bs1) -> bf16
  gemm_wmma_kernel<false, true, true><<<gHid, blk, 0, stream>>>(x, Ws1, bs1, Hb, N_, HID_, D_);

  // importance score + top-K selection mask
  imp_kernel<<<N_ / 8, 256, 0, stream>>>(Hb, Ws2, bs2, imp);
  topk_kernel<<<B_, 256, 0, stream>>>(imp, sel);

  // masked flash attention -> ATTb (bf16, [B,S,D])
  attn_kernel<<<dim3(S_ / 64, B_ * H_), blk, 0, stream>>>(Qb, Kb, Vb, sel, ATTb);

  // output projection (bf16 A -> fp32 out): d_out = ATTb @ Wo + bo
  gemm_wmma_kernel<true, false, false><<<gProj, blk, 0, stream>>>(ATTb, Wo, bo, (float*)d_out, N_, D_, D_);
}
